// VanillaRNN_78108275245324
// MI455X (gfx1250) — compile-verified, matrix operand order checked by reading, not run on hardware
//
#include <hip/hip_runtime.h>
#include <stdint.h>

// Problem dims (fixed by the reference)
#define B    256
#define S    512
#define IDIM 256
#define H    1024
#define ODIM 128

#define THREADS 256          // 8 waves per block (wave32)
#define RNN_NWG 64           // persistent grid: 64 blocks * 8 waves = 512 waves

typedef __attribute__((ext_vector_type(16))) __bf16 v16bf;
typedef __attribute__((ext_vector_type(8)))  float  v8f;

__device__ __forceinline__ unsigned short f32_to_bf16(float f) {
  unsigned u = __float_as_uint(f);
  u += 0x7FFFu + ((u >> 16) & 1u);      // round-to-nearest-even
  return (unsigned short)(u >> 16);
}

// Fast tanh: stable at +/-inf (exp overflow -> 1, underflow -> -1)
__device__ __forceinline__ float fast_tanh(float x) {
  float e = __expf(2.0f * x);
  return 1.0f - 2.0f / (e + 1.0f);
}

// Load a 16x32 bf16 WMMA fragment slice for this lane (row-major source).
// ISA layout (CDNA5 ISA §7.12.2): lane L holds row M=L%16,
// K = (L/16)*8..+7 in VGPRs 0-3 and K = 16+(L/16)*8..+7 in VGPRs 4-7.
// Works for global (global_load_b128) and LDS (ds_load_b128) pointers.
__device__ __forceinline__ v16bf load_frag16(const unsigned short* p) {
  union { uint4 u[2]; v16bf v; } t;
  t.u[0] = *reinterpret_cast<const uint4*>(p);
  t.u[1] = *reinterpret_cast<const uint4*>(p + 16);
  return t.v;
}

// ---------------------------------------------------------------------------
// Kernel 1: convert f32 inputs to bf16 workspace, zero h0 and the sync counter
// ---------------------------------------------------------------------------
__global__ void convert_kernel(const float* __restrict__ x,
                               const float* __restrict__ whx,
                               const float* __restrict__ whh,
                               const float* __restrict__ who,
                               unsigned short* __restrict__ xb,
                               unsigned short* __restrict__ whxb,
                               unsigned short* __restrict__ whhb,
                               unsigned short* __restrict__ whob,
                               unsigned short* __restrict__ h0,
                               unsigned* __restrict__ cnt) {
  size_t g = (size_t)blockIdx.x * blockDim.x + threadIdx.x;
  size_t stride = (size_t)gridDim.x * blockDim.x;
  const size_t XN = (size_t)B * S * IDIM;
  for (size_t i = g; i < XN; i += stride)                 xb[i]   = f32_to_bf16(x[i]);
  for (size_t i = g; i < (size_t)H * IDIM; i += stride)   whxb[i] = f32_to_bf16(whx[i]);
  for (size_t i = g; i < (size_t)H * H; i += stride)      whhb[i] = f32_to_bf16(whh[i]);
  for (size_t i = g; i < (size_t)ODIM * H; i += stride)   whob[i] = f32_to_bf16(who[i]);
  for (size_t i = g; i < (size_t)B * H; i += stride)      h0[i]   = 0;
  if (g == 0) *cnt = 0u;
}

// ---------------------------------------------------------------------------
// Kernel 2: persistent recurrent scan.
//   h_new = tanh(x_t @ W_hx^T + b_hx + h @ W_hh^T)
// Block = 8 row-tiles x 2 col-tiles; all 8 waves share the 2 col-tiles whose
// 32 W_hh rows (64 KB bf16) are staged into LDS once via async load-to-LDS
// and read each step with ds_load_b128. W_hx fragments are loop-invariant
// (VGPR-resident). Per step: x A-fragments are issued FIRST into registers
// (one 16-deep clause), the long h-part then hides their HBM latency, and
// the x WMMAs run last. The x row two steps ahead is prefetched. Time loop is
// unrolled by 2 so h ping-pong pointers are static. Grid barrier per step.
// ---------------------------------------------------------------------------
__global__ void __launch_bounds__(THREADS, 1)
rnn_scan_kernel(const unsigned short* __restrict__ xb,    // [B][S][IDIM] bf16
                const unsigned short* __restrict__ whxb,  // [H][IDIM]    bf16
                const unsigned short* __restrict__ whhb,  // [H][H]       bf16
                const float* __restrict__ b_hx,           // [H]          f32
                unsigned short* __restrict__ h0,          // [B][H] bf16 ping
                unsigned short* __restrict__ h1,          // [B][H] bf16 pong
                unsigned* __restrict__ cnt) {
  const int lane = threadIdx.x & 31;
  const int wave = threadIdx.x >> 5;

  // Tiling: 16 row-tiles (B/16) x 64 col-tiles (H/16).
  const int np     = blockIdx.x >> 1;          // col-tile pair {2np, 2np+1}
  const int mgroup = blockIdx.x & 1;           // row-tile half
  const int mt     = mgroup * 8 + wave;        // 0..15
  const int nbase  = np * 32;                  // first W_hh row staged in LDS

  const int mrow = mt * 16 + (lane & 15);      // A-matrix row for this lane
  const int kb   = (lane >> 4) << 3;           // lane-half K offset (0 or 8)
  const int n0   = nbase + (lane & 15);        // output columns
  const int n1   = nbase + 16 + (lane & 15);

  const float bias0 = b_hx[n0];
  const float bias1 = b_hx[n1];

  // C/D layout: VGPR r -> M = r (lanes 0-15) or r+8 (lanes 16-31), N = lane&15
  const int rbase = mt * 16 + ((lane >> 4) << 3);

  // ---- Stage the block's 32 W_hh rows into LDS (64 KB), once, async ----
  __shared__ __align__(16) unsigned short wlds[32 * H];
  {
    const unsigned short* src = whhb + (size_t)nbase * H;
    for (int i = threadIdx.x; i < (32 * H) / 8; i += THREADS) {  // 16B chunks
      unsigned lds_off = (unsigned)(uintptr_t)(&wlds[i * 8]);
      const unsigned short* g = src + (size_t)i * 8;
      asm volatile("global_load_async_to_lds_b128 %0, %1, off"
                   :: "v"(lds_off), "v"(g) : "memory");
    }
    asm volatile("s_wait_asynccnt 0x0" ::: "memory");
    __syncthreads();
  }

  // Per-lane LDS fragment base pointers (local rows lane&15 and 16+(lane&15))
  const unsigned short* wl0 = &wlds[(size_t)(lane & 15) * H + kb];
  const unsigned short* wl1 = &wlds[(size_t)(16 + (lane & 15)) * H + kb];

  const unsigned short* xrow_base = xb + (size_t)mrow * (S * IDIM);
  const unsigned short* wx0 = whxb + (size_t)n0 * IDIM + kb;
  const unsigned short* wx1 = whxb + (size_t)n1 * IDIM + kb;

  // One recurrent step: read hc, write hn (both statically known pointers)
  auto step = [&](int t, const unsigned short* __restrict__ hc,
                  unsigned short* __restrict__ hn) {
    // 1) Issue all x A-fragment loads up front (distinct registers -> one
    //    clause of 16 global_load_b128); consumed only after the h-part.
    const unsigned short* xrow = xrow_base + (size_t)t * IDIM + kb;
    v16bf ax[IDIM / 32];
#pragma unroll
    for (int i = 0; i < IDIM / 32; ++i) ax[i] = load_frag16(xrow + i * 32);

    // Prefetch the x row two steps ahead (t+2) into L2 while we compute this one.
    {
      int tp = (t + 2 < S) ? (t + 2) : (S - 1);
      __builtin_prefetch(xrow_base + (size_t)tp * IDIM, 0, 1);
    }

    v8f c0 = {}; v8f c1 = {};

    // 2) h-part: K over H; A from global (L2-hot), B from LDS. This long
    //    section hides the HBM latency of the x loads issued above.
    const unsigned short* hrow = hc + (size_t)mrow * H + kb;
#pragma unroll 8
    for (int kk = 0; kk < H; kk += 32) {
      v16bf a  = load_frag16(hrow + kk);
      v16bf bA = load_frag16(wl0 + kk);
      v16bf bB = load_frag16(wl1 + kk);
      c0 = __builtin_amdgcn_wmma_f32_16x16x32_bf16(false, a, false, bA, (short)0, c0, false, false);
      c1 = __builtin_amdgcn_wmma_f32_16x16x32_bf16(false, a, false, bB, (short)0, c1, false, false);
    }

    // 3) x-part WMMAs from registers; W_hx fragments are loop-invariant and
    //    hoisted into VGPRs by the compiler.
#pragma unroll
    for (int i = 0; i < IDIM / 32; ++i) {
      v16bf bA = load_frag16(wx0 + i * 32);
      v16bf bB = load_frag16(wx1 + i * 32);
      c0 = __builtin_amdgcn_wmma_f32_16x16x32_bf16(false, ax[i], false, bA, (short)0, c0, false, false);
      c1 = __builtin_amdgcn_wmma_f32_16x16x32_bf16(false, ax[i], false, bB, (short)0, c1, false, false);
    }

    // 4) epilogue: bias + tanh, store bf16 h_next
#pragma unroll
    for (int r = 0; r < 8; ++r) {
      const size_t row = (size_t)(rbase + r) * H;
      hn[row + n0] = f32_to_bf16(fast_tanh(c0[r] + bias0));
      hn[row + n1] = f32_to_bf16(fast_tanh(c1[r] + bias1));
    }
  };

  // Grid-wide barrier via monotonic counter
  auto gsync = [&](unsigned target) {
    __threadfence();
    __syncthreads();
    if (threadIdx.x == 0) {
      __hip_atomic_fetch_add(cnt, 1u, __ATOMIC_ACQ_REL, __HIP_MEMORY_SCOPE_AGENT);
      while (__hip_atomic_load(cnt, __ATOMIC_ACQUIRE, __HIP_MEMORY_SCOPE_AGENT) < target) {
        __builtin_amdgcn_s_sleep(2);
      }
    }
    __syncthreads();
  };

  for (int t = 0; t < S; t += 2) {
    step(t,     h0, h1);
    gsync((unsigned)(t + 1) * RNN_NWG);
    step(t + 1, h1, h0);
    gsync((unsigned)(t + 2) * RNN_NWG);
  }
}

// ---------------------------------------------------------------------------
// Kernel 3: logits = h_last @ W_ho^T + b_ho  (256x128, K=1024) via WMMA
// ---------------------------------------------------------------------------
__global__ void __launch_bounds__(THREADS)
out_gemm_kernel(const unsigned short* __restrict__ hlast,  // [B][H] bf16
                const unsigned short* __restrict__ whob,   // [ODIM][H] bf16
                const float* __restrict__ b_ho,            // [ODIM]
                float* __restrict__ logits) {              // [B][ODIM]
  const int lane = threadIdx.x & 31;
  const int wave = threadIdx.x >> 5;
  const int gw   = blockIdx.x * (THREADS / 32) + wave;  // 0..127
  const int mt   = gw & 15;   // 16 row tiles
  const int nt   = gw >> 4;   // 8 col tiles

  const int mrow = mt * 16 + (lane & 15);
  const int kb   = (lane >> 4) << 3;
  const int n    = nt * 16 + (lane & 15);

  v8f c = {};
  const unsigned short* ha = hlast + (size_t)mrow * H + kb;
  const unsigned short* wb = whob + (size_t)n * H + kb;
#pragma unroll 4
  for (int kk = 0; kk < H; kk += 32) {
    v16bf a = load_frag16(ha + kk);
    v16bf b = load_frag16(wb + kk);
    c = __builtin_amdgcn_wmma_f32_16x16x32_bf16(false, a, false, b, (short)0, c, false, false);
  }
  const float bias = b_ho[n];
  const int rbase = mt * 16 + ((lane >> 4) << 3);
#pragma unroll
  for (int r = 0; r < 8; ++r)
    logits[(size_t)(rbase + r) * ODIM + n] = c[r] + bias;
}

// ---------------------------------------------------------------------------
// Kernel 4: row softmax over 128 columns, one wave32 per row
// ---------------------------------------------------------------------------
__global__ void softmax_kernel(const float* __restrict__ logits,
                               float* __restrict__ out) {
  const int lane = threadIdx.x & 31;
  const int wave = threadIdx.x >> 5;
  const int row  = blockIdx.x * (blockDim.x >> 5) + wave;  // exact coverage

  const float* lg = logits + (size_t)row * ODIM;
  float v[4];
#pragma unroll
  for (int i = 0; i < 4; ++i) v[i] = lg[lane + 32 * i];

  float m = fmaxf(fmaxf(v[0], v[1]), fmaxf(v[2], v[3]));
#pragma unroll
  for (int off = 16; off > 0; off >>= 1) m = fmaxf(m, __shfl_xor(m, off, 32));

  float s = 0.f;
#pragma unroll
  for (int i = 0; i < 4; ++i) { v[i] = __expf(v[i] - m); s += v[i]; }
#pragma unroll
  for (int off = 16; off > 0; off >>= 1) s += __shfl_xor(s, off, 32);

  const float inv = 1.0f / s;
#pragma unroll
  for (int i = 0; i < 4; ++i) out[(size_t)row * ODIM + lane + 32 * i] = v[i] * inv;
}

// ---------------------------------------------------------------------------
extern "C" void kernel_launch(void* const* d_in, const int* in_sizes, int n_in,
                              void* d_out, int out_size, void* d_ws, size_t ws_size,
                              hipStream_t stream) {
  (void)in_sizes; (void)n_in; (void)out_size; (void)ws_size;
  const float* x   = (const float*)d_in[0];   // [B][S][IDIM]
  const float* Whx = (const float*)d_in[1];   // [H][IDIM]
  const float* bhx = (const float*)d_in[2];   // [H]
  const float* Whh = (const float*)d_in[3];   // [H][H]
  const float* Who = (const float*)d_in[4];   // [ODIM][H]
  const float* bho = (const float*)d_in[5];   // [ODIM]
  float* out = (float*)d_out;                 // [B][ODIM] f32

  // Workspace carve-out (~71 MB total)
  char* ws = (char*)d_ws;
  size_t off = 0;
  auto carve = [&](size_t bytes) -> char* {
    char* p = ws + off;
    off += bytes;
    off = (off + 255) & ~(size_t)255;
    return p;
  };
  unsigned short* xb     = (unsigned short*)carve((size_t)B * S * IDIM * 2);
  unsigned short* whxb   = (unsigned short*)carve((size_t)H * IDIM * 2);
  unsigned short* whhb   = (unsigned short*)carve((size_t)H * H * 2);
  unsigned short* whob   = (unsigned short*)carve((size_t)ODIM * H * 2);
  unsigned short* h0     = (unsigned short*)carve((size_t)B * H * 2);
  unsigned short* h1     = (unsigned short*)carve((size_t)B * H * 2);
  float*          logits = (float*)carve((size_t)B * ODIM * 4);
  unsigned*       cnt    = (unsigned*)carve(256);

  convert_kernel<<<2048, THREADS, 0, stream>>>(x, Whx, Whh, Who,
                                               xb, whxb, whhb, whob, h0, cnt);

  rnn_scan_kernel<<<RNN_NWG, THREADS, 0, stream>>>(xb, whxb, whhb, bhx,
                                                   h0, h1, cnt);

  // S = 512 is even -> final hidden state lands back in h0
  out_gemm_kernel<<<16, THREADS, 0, stream>>>(h0, whob, bho, logits);

  softmax_kernel<<<B / (THREADS / 32), THREADS, 0, stream>>>(logits, out);
}
